// ChebyKANLinear_60610578481558
// MI455X (gfx1250) — matrix operand order repaired and checked by reading, not compiled, not run on hardware
//
#include <hip/hip_runtime.h>

typedef __attribute__((ext_vector_type(16))) _Float16 v16h;
typedef __attribute__((ext_vector_type(8)))  float    v8f;
typedef __attribute__((ext_vector_type(4)))  float    f32x4;
typedef __attribute__((ext_vector_type(4)))  unsigned int u32x4;
typedef __attribute__((ext_vector_type(2)))  unsigned int u32x2;

#define BATCH  8192
#define IN_F   1024
#define OUT_F  1024
#define NDEG   9                 // D+1
#define KTOT   (IN_F * NDEG)     // 9216

#define BM 128
#define BN 128
#define LDA 40                   // halves per LDS row (32 data + 8 pad, 80B stride: 16B-aligned)
#define BUFH (BM * LDA)          // halves per LDS buffer (10240 B)

// ---------------------------------------------------------------------------
// Prep: coef (I, O, 9) fp32  ->  Wt (O, K) fp16 with K = d*1024 + i.
// Reads fully coalesced; scattered fp16 writes land in the L2-resident 18MB Wt.
// ---------------------------------------------------------------------------
__global__ __launch_bounds__(256) void cheby_prep(const float* __restrict__ coef,
                                                  _Float16* __restrict__ wt) {
    int tid  = blockIdx.x * blockDim.x + threadIdx.x;   // = i*(O*9) + o*9 + d
    int d    = tid % NDEG;
    int rest = tid / NDEG;
    int o    = rest % OUT_F;
    int i    = rest / OUT_F;
    wt[(size_t)o * KTOT + d * IN_F + i] = (_Float16)coef[tid];
}

// ---------------------------------------------------------------------------
// Fused ChebyKAN GEMM, double-buffered LDS, compile-time ping-pong addressing.
// out[b,o] = sum_{i,d} T_d(tanh(x[b,i])) * W[o, d*1024+i]
// ---------------------------------------------------------------------------
__global__ __launch_bounds__(256) void cheby_gemm(const float* __restrict__ x,
                                                  const _Float16* __restrict__ wt,
                                                  float* __restrict__ out) {
    __shared__ _Float16 As[2 * BUFH];   // feature tiles (T_d), ping-pong
    __shared__ _Float16 Ws[2 * BUFH];   // weight tiles (column-major in k), ping-pong

    const int tid   = threadIdx.x;
    const int lane  = tid & 31;
    const int wave  = tid >> 5;       // 8 waves
    const int waveM = wave & 3;       // 4 waves along M (32 rows each)
    const int waveN = wave >> 2;      // 2 waves along N (64 cols each)

    const int m0 = blockIdx.y * BM;
    const int n0 = blockIdx.x * BN;

    // x-tile ownership: thread owns rows xr, xr+32, xr+64, xr+96; cols xg*4..xg*4+3
    const int xr = tid >> 3;          // 0..31
    const int xg = tid & 7;           // 0..7

    // Ws staging: thread loads 32B (16 halves) of row wrow
    const int wrow = tid >> 1;            // 0..127
    const int woff = (tid & 1) * 16;      // 0 or 16 halves

    const float*    xbase  = x + (size_t)(m0 + xr) * IN_F + xg * 4;
    const _Float16* wtbase = wt + (size_t)(n0 + wrow) * KTOT + woff;

    float tq[16], t2[16], tm1[16], tm2[16];

    v8f acc[2][4];
#pragma unroll
    for (int q = 0; q < 2; ++q)
#pragma unroll
        for (int p = 0; p < 4; ++p) acc[q][p] = (v8f){};

    // fragment addressing (ISA 7.12.2 f16 layouts), constant across K loop
    const int asel  = (lane >= 16) ? 8 : 0;     // A: half-column offset
    const int al    = lane & 15;
    const int bks   = (lane >= 16) ? 8 : 0;     // B: k offset
    const int aIdx  = (waveM * 32 + al) * LDA + asel;
    const int bIdx  = (waveN * 64 + al) * LDA + bks;
    const int wsIdx = wrow * LDA + woff;
    const int asOut = xr * LDA + xg * 4;        // base of this thread's As stores

    // ---- prologue: prepare step (ic=0, d=0) into buffer 0 ----
    {
#pragma unroll
        for (int q = 0; q < 4; ++q) {
            f32x4 v = *(const f32x4*)(xbase + (size_t)q * 32 * IN_F);
#pragma unroll
            for (int c = 0; c < 4; ++c) tq[q * 4 + c] = tanhf(v[c]);
        }
#pragma unroll
        for (int j = 0; j < 16; ++j) {
            t2[j] = tq[j] + tq[j];
            tm2[j] = 1.0f;
            tm1[j] = tq[j];
        }
        // T_0 == 1 feature tile
#pragma unroll
        for (int q = 0; q < 4; ++q) {
            union { _Float16 h[4]; u32x2 u; } pk;
#pragma unroll
            for (int c = 0; c < 4; ++c) pk.h[c] = (_Float16)1.0f;
            *(u32x2*)&As[asOut + q * 32 * LDA] = pk.u;
        }
        u32x4 wa = *(const u32x4*)wtbase;        // (d=0, i0=0) chunk
        u32x4 wb = *(const u32x4*)(wtbase + 8);
        *(u32x4*)&Ws[wsIdx]     = wa;
        *(u32x4*)&Ws[wsIdx + 8] = wb;
        __syncthreads();
    }

    // i-chunks two at a time so buffer parity is compile-time (9 steps = odd flip)
    for (int ic2 = 0; ic2 < IN_F / 64; ++ic2) {
#pragma unroll
        for (int half = 0; half < 2; ++half) {
            const int ic = ic2 * 2 + half;
#pragma unroll
            for (int d = 0; d < NDEG; ++d) {
                const int  bofs  = ((half + d) & 1) * BUFH;   // compile-time
                const int  nbofs = bofs ^ BUFH;               // compile-time
                const bool lastStep = (half == 1) && (d == NDEG - 1) &&
                                      (ic2 == IN_F / 64 - 1);
                const int  nd  = (d < NDEG - 1) ? d + 1 : 0;  // next degree
                const int  ni0 = (d < NDEG - 1) ? ic * 32 : (ic + 1) * 32;

                // (1) issue next weight-chunk global load
                u32x4 wa = {}, wb = {};
                if (!lastStep) {
                    const _Float16* wsrc = wtbase + nd * IN_F + ni0;
                    wa = *(const u32x4*)wsrc;
                    wb = *(const u32x4*)(wsrc + 8);
                }

                // (2) advance Chebyshev recursion for the NEXT tile
                float cur[16];
                if (!lastStep) {
                    if (nd == 0) {        // new i-chunk: reload x, tanh, T_0 = 1
#pragma unroll
                        for (int q = 0; q < 4; ++q) {
                            f32x4 v = *(const f32x4*)(xbase + (size_t)q * 32 * IN_F + ni0);
#pragma unroll
                            for (int c = 0; c < 4; ++c) tq[q * 4 + c] = tanhf(v[c]);
                        }
#pragma unroll
                        for (int j = 0; j < 16; ++j) {
                            t2[j] = tq[j] + tq[j];
                            tm2[j] = 1.0f;
                            tm1[j] = tq[j];
                            cur[j] = 1.0f;
                        }
                    } else if (nd == 1) { // T_1 = t
#pragma unroll
                        for (int j = 0; j < 16; ++j) cur[j] = tq[j];
                    } else {              // T_k = (2t)·T_{k-1} - T_{k-2}  (t2 hoisted)
#pragma unroll
                        for (int j = 0; j < 16; ++j) {
                            float Tc = t2[j] * tm1[j] - tm2[j];
                            tm2[j] = tm1[j];
                            tm1[j] = Tc;
                            cur[j] = Tc;
                        }
                    }
                }

                // (3) fragments from CURRENT buffer + 8 WMMAs
                v16h afrag[2];
#pragma unroll
                for (int q = 0; q < 2; ++q) {
                    const _Float16* p = &As[bofs + aIdx + q * 16 * LDA];
                    union { u32x4 u[2]; v16h h; } f;
                    f.u[0] = *(const u32x4*)p;          // K 0..7   (or 8..15)
                    f.u[1] = *(const u32x4*)(p + 16);   // K 16..23 (or 24..31)
                    afrag[q] = f.h;
                }
                v16h bfrag[4];
#pragma unroll
                for (int p = 0; p < 4; ++p) {
                    const _Float16* q = &Ws[bofs + bIdx + p * 16 * LDA];
                    union { u32x4 u[2]; v16h h; } f;
                    f.u[0] = *(const u32x4*)q;          // K 0..7   (or 8..15)
                    f.u[1] = *(const u32x4*)(q + 16);   // K 16..23 (or 24..31)
                    bfrag[p] = f.h;
                }
#pragma unroll
                for (int q = 0; q < 2; ++q)
#pragma unroll
                    for (int p = 0; p < 4; ++p)
                        acc[q][p] = __builtin_amdgcn_wmma_f32_16x16x32_f16(
                            false, afrag[q], false, bfrag[p],
                            (short)0, acc[q][p], false, false);

                // (4) write NEXT tiles into the other buffer (immediate DS offsets)
                if (!lastStep) {
#pragma unroll
                    for (int q = 0; q < 4; ++q) {
                        union { _Float16 h[4]; u32x2 u; } pk;
#pragma unroll
                        for (int c = 0; c < 4; ++c) pk.h[c] = (_Float16)cur[q * 4 + c];
                        *(u32x2*)&As[nbofs + asOut + q * 32 * LDA] = pk.u;
                    }
                    *(u32x4*)&Ws[nbofs + wsIdx]     = wa;
                    *(u32x4*)&Ws[nbofs + wsIdx + 8] = wb;
                }

                // (5) single barrier: publishes buf^1, retires reads of buf
                __syncthreads();
            }
        }
    }

    // ---- epilogue: C/D layout — VGPR v holds (M = v + 8*(lane>=16), N = lane&15) ----
    const int orow = (lane >= 16) ? 8 : 0;
    const int ocol = lane & 15;
#pragma unroll
    for (int q = 0; q < 2; ++q)
#pragma unroll
        for (int p = 0; p < 4; ++p) {
            float* op = out + (size_t)(m0 + waveM * 32 + q * 16 + orow) * OUT_F
                            + n0 + waveN * 64 + p * 16 + ocol;
#pragma unroll
            for (int v = 0; v < 8; ++v)
                op[(size_t)v * OUT_F] = acc[q][p][v];
        }
}

// ---------------------------------------------------------------------------
extern "C" void kernel_launch(void* const* d_in, const int* in_sizes, int n_in,
                              void* d_out, int out_size, void* d_ws, size_t ws_size,
                              hipStream_t stream) {
    const float* x    = (const float*)d_in[0];   // (8192, 1024) f32
    const float* coef = (const float*)d_in[1];   // (1024, 1024, 9) f32
    float* out        = (float*)d_out;           // (8192, 1024) f32
    _Float16* wt      = (_Float16*)d_ws;         // (1024, 9216) f16 = 18 MB

    const int total = IN_F * OUT_F * NDEG;       // 9,437,184
    cheby_prep<<<total / 256, 256, 0, stream>>>(coef, wt);

    dim3 grid(OUT_F / BN, BATCH / BM);           // (8, 64)
    cheby_gemm<<<grid, 256, 0, stream>>>(x, wt, out);
}
